// FNO_62766652064096
// MI455X (gfx1250) — matrix operand order repaired and checked by reading, not compiled, not run on hardware
//
#include <hip/hip_runtime.h>
#include <hip/hip_bf16.h>
#include <math.h>

typedef __attribute__((ext_vector_type(16))) _Float16 v16h;
typedef __attribute__((ext_vector_type(8)))  _Float16 v8h;
typedef __attribute__((ext_vector_type(4)))  _Float16 v4h;
typedef __attribute__((ext_vector_type(8)))  float    v8f;
typedef int v4i_vs __attribute__((vector_size(4 * sizeof(int))));  // matches builtin param type

#define B_    8
#define H_    128
#define W_    128
#define CIN_  4
#define C_    64
#define KMAX_ 16
#define STAGES_ 6
#define PH_   128
#define NPIX_ (B_*H_*W_)   // 131072

// gfx1250 async global->LDS path (guarded; falls back to plain loads)
#if defined(__has_builtin)
#if __has_builtin(__builtin_amdgcn_global_load_async_to_lds_b128) && \
    __has_builtin(__builtin_amdgcn_s_wait_asynccnt)
#define HAVE_GFX1250_ASYNC 1
#endif
#endif

__device__ __forceinline__ float gelu_approx(float x) {
    const float c0 = 0.7978845608028654f; // sqrt(2/pi)
    float x3 = x * x * x;
    float t  = tanhf(c0 * (x + 0.044715f * x3));
    return 0.5f * x * (1.0f + t);
}

// Build A fragment (16-bit A 16x32 layout) from a row-major [M][K] f16 LDS row.
// lanes<16 hold K kc+0..15; lanes>=16 hold K kc+16..31.
// One contiguous 16-half (32B) run.
__device__ __forceinline__ v16h frag_a(const _Float16* row, int kc, int lane) {
    int abase = kc + ((lane & 16) ? 16 : 0);
    return *(const v16h*)(row + abase);
}

// Build B fragment (16-bit B 32x16 layout) from an N-major [N][K] f16 LDS row.
// lane -> N = lane&15; lanes<16 hold K kc+0..15, lanes>=16 hold K kc+16..31:
// one contiguous 16-half (32B) run.
__device__ __forceinline__ v16h frag_b(const _Float16* rowN, int kc, int lane) {
    int bbase = kc + ((lane & 16) ? 16 : 0);
    return *(const v16h*)(rowN + bbase);
}

// ---------------------------------------------------------------- twiddles
__global__ void k_twiddle(float* twc, float* tws) {
    int i = blockIdx.x * blockDim.x + threadIdx.x;
    if (i < KMAX_ * H_) {
        int ki = i / H_, h = i % H_;
        float ang = 6.283185307179586f * (float)(ki * h) / (float)H_;
        twc[i] = cosf(ang);
        tws[i] = sinf(ang);
    }
}

// ---------------------------------------------------------------- lift 4->64 (coalesced)
__global__ void k_lift(const float* __restrict__ x, const float* __restrict__ lw,
                       const float* __restrict__ lb, float* __restrict__ hout) {
    int idx = blockIdx.x * blockDim.x + threadIdx.x;
    if (idx >= NPIX_ * C_) return;
    int c = idx % C_;
    int p = idx / C_;
    float a = lb[c];
#pragma unroll
    for (int i = 0; i < CIN_; i++) a += x[p * CIN_ + i] * lw[i * C_ + c];
    hout[idx] = a;
}

// ---------------------------------------------------------------- row DFT (axis 1)
// R[b,ki,w,c] = sum_h hin[b,h,w,c] * exp(-2pi i ki h / H)
__global__ void k_rowdft(const float* __restrict__ hin,
                         const float* __restrict__ twc, const float* __restrict__ tws,
                         float* __restrict__ Rre, float* __restrict__ Rim) {
    int idx = blockIdx.x * blockDim.x + threadIdx.x;
    if (idx >= B_ * W_ * C_) return;
    int c = idx % C_;
    int w = (idx / C_) % W_;
    int b = idx / (C_ * W_);
    // warm the far half of this column (emits global_prefetch_b8)
    __builtin_prefetch(&hin[((b * H_ + 64) * W_ + w) * C_ + c], 0, 1);
    float are[KMAX_], aim[KMAX_];
#pragma unroll
    for (int k = 0; k < KMAX_; k++) { are[k] = 0.f; aim[k] = 0.f; }
    for (int h = 0; h < H_; h++) {
        float v = hin[((b * H_ + h) * W_ + w) * C_ + c];
#pragma unroll
        for (int k = 0; k < KMAX_; k++) {
            are[k] += v * twc[k * H_ + h];
            aim[k] -= v * tws[k * H_ + h];
        }
    }
#pragma unroll
    for (int k = 0; k < KMAX_; k++) {
        int o = ((b * KMAX_ + k) * W_ + w) * C_ + c;
        Rre[o] = are[k]; Rim[o] = aim[k];
    }
}

// ---------------------------------------------------------------- col DFT (axis 2)
__global__ void k_coldft(const float* __restrict__ Rre, const float* __restrict__ Rim,
                         const float* __restrict__ twc, const float* __restrict__ tws,
                         float* __restrict__ Lre, float* __restrict__ Lim) {
    int idx = blockIdx.x * blockDim.x + threadIdx.x;
    if (idx >= B_ * KMAX_ * KMAX_ * C_) return;
    int c  = idx % C_;
    int kj = (idx / C_) % KMAX_;
    int ki = (idx / (C_ * KMAX_)) % KMAX_;
    int b  = idx / (C_ * KMAX_ * KMAX_);
    float sre = 0.f, sim = 0.f;
    for (int w = 0; w < W_; w++) {
        float cr = twc[kj * W_ + w], si = tws[kj * W_ + w];
        int o = ((b * KMAX_ + ki) * W_ + w) * C_ + c;
        float xr = Rre[o], xi = Rim[o];
        sre += xr * cr + xi * si;       // (xr + i xi) * (cr - i si)
        sim += xi * cr - xr * si;
    }
    int o = ((b * KMAX_ + ki) * KMAX_ + kj) * C_ + c;
    Lre[o] = sre; Lim[o] = sim;
}

// ---------------------------------------------------------------- spectral mixing
__global__ void k_mix(const float* __restrict__ Lre, const float* __restrict__ Lim,
                      const float* __restrict__ kern,
                      float* __restrict__ Mre, float* __restrict__ Mim) {
    int idx = blockIdx.x * blockDim.x + threadIdx.x;
    if (idx >= B_ * KMAX_ * KMAX_ * C_) return;
    int o  = idx % C_;
    int kj = (idx / C_) % KMAX_;
    int ki = (idx / (C_ * KMAX_)) % KMAX_;
    int b  = idx / (C_ * KMAX_ * KMAX_);
    float sre = 0.f, sim = 0.f;
    int lbase = ((b * KMAX_ + ki) * KMAX_ + kj) * C_;
    for (int c = 0; c < C_; c++) {
        float xr = Lre[lbase + c], xi = Lim[lbase + c];
        const float* kp = kern + (size_t)(((c * C_ + o) * KMAX_ + ki) * KMAX_ + kj) * 2;
        float wr = kp[0], wi = kp[1];
        sre += xr * wr - xi * wi;
        sim += xr * wi + xi * wr;
    }
    int od = ((b * KMAX_ + ki) * KMAX_ + kj) * C_ + o;
    Mre[od] = sre; Mim[od] = sim;
}

// ---------------------------------------------------------------- inverse row
__global__ void k_irow(const float* __restrict__ Mre, const float* __restrict__ Mim,
                       const float* __restrict__ twc, const float* __restrict__ tws,
                       float* __restrict__ Zre, float* __restrict__ Zim) {
    int idx = blockIdx.x * blockDim.x + threadIdx.x;
    if (idx >= B_ * H_ * KMAX_ * C_) return;
    int c  = idx % C_;
    int kj = (idx / C_) % KMAX_;
    int h  = (idx / (C_ * KMAX_)) % H_;
    int b  = idx / (C_ * KMAX_ * H_);
    float sre = 0.f, sim = 0.f;
#pragma unroll
    for (int ki = 0; ki < KMAX_; ki++) {
        float cr = twc[ki * H_ + h], si = tws[ki * H_ + h];
        int o = ((b * KMAX_ + ki) * KMAX_ + kj) * C_ + c;
        float xr = Mre[o], xi = Mim[o];
        sre += xr * cr - xi * si;       // * (cr + i si)
        sim += xr * si + xi * cr;
    }
    int o = ((b * H_ + h) * KMAX_ + kj) * C_ + c;
    Zre[o] = sre; Zim[o] = sim;
}

// ---------------------------------------------------------------- inverse col -> filt
__global__ void k_icol(const float* __restrict__ Zre, const float* __restrict__ Zim,
                       const float* __restrict__ twc, const float* __restrict__ tws,
                       float* __restrict__ filt) {
    int idx = blockIdx.x * blockDim.x + threadIdx.x;
    if (idx >= NPIX_ * C_) return;
    int c = idx % C_;
    int w = (idx / C_) % W_;
    int h = (idx / (C_ * W_)) % H_;
    int b = idx / (C_ * W_ * H_);
    int zb = ((b * H_ + h) * KMAX_) * C_ + c;
    float acc = Zre[zb];  // kj = 0
#pragma unroll
    for (int kj = 1; kj < KMAX_; kj++) {
        float cr = twc[kj * W_ + w], si = tws[kj * W_ + w];
        acc += 2.0f * (Zre[zb + kj * C_] * cr - Zim[zb + kj * C_] * si);
    }
    filt[idx] = acc * (1.0f / (float)(H_ * W_));
}

// ---------------------------------------------------------------- 1x1 conv, WMMA f16 16x16x32
// out[p,o] = gelu( sum_c hin[p,c]*wmat[c,o] + bias[o] + filt[p,o] );  filt may alias hout.
// filt tile is fetched with async global->LDS (overlapped with the GEMM) when available.
__global__ __launch_bounds__(256) void k_conv_wmma(
        const float* __restrict__ hin, const float* __restrict__ filt,
        const float* __restrict__ wmat, const float* __restrict__ bias,
        float* __restrict__ hout) {
    __shared__ __align__(32) _Float16 Ash[32][C_];    // [M][K] f16
    __shared__ __align__(32) _Float16 BshT[C_][C_];   // [N][K] f16 (transposed)
    __shared__ __align__(16) float    Flds[32][C_];   // filt tile f32
    int tid = threadIdx.x;
    int pbase = blockIdx.x * 32;

    // filt tile: 32x64 f32 = 512 float4 -> async copy to LDS, overlapped with GEMM
#if defined(HAVE_GFX1250_ASYNC)
    for (int i = tid; i < (32 * C_) / 4; i += 256) {
        int m = (i * 4) / C_, k0 = (i * 4) % C_;
        __builtin_amdgcn_global_load_async_to_lds_b128(
            (v4i_vs*)(filt + (size_t)(pbase + m) * C_ + k0),
            (v4i_vs*)(&Flds[m][k0]), 0, 0);
    }
#else
    for (int i = tid; i < (32 * C_) / 4; i += 256) {
        int m = (i * 4) / C_, k0 = (i * 4) % C_;
        *(float4*)&Flds[m][k0] = *(const float4*)(filt + (size_t)(pbase + m) * C_ + k0);
    }
#endif
    // A tile: float4 loads + f16 convert
    for (int i = tid; i < (32 * C_) / 4; i += 256) {
        int m = (i * 4) / C_, k0 = (i * 4) % C_;
        float4 v = *(const float4*)(hin + (size_t)(pbase + m) * C_ + k0);
        v4h h4 = { (_Float16)v.x, (_Float16)v.y, (_Float16)v.z, (_Float16)v.w };
        *(v4h*)&Ash[m][k0] = h4;
    }
    // B tile transposed: BshT[n][k] = wmat[k][n]
    for (int i = tid; i < C_ * C_; i += 256) {
        int k = i / C_, n = i % C_;
        BshT[n][k] = (_Float16)wmat[i];
    }
    __syncthreads();

    int wave = tid >> 5, lane = tid & 31;
    int mTile = wave >> 2, nTile = wave & 3;   // 2 M-tiles x 4 N-tiles
    int l15 = lane & 15;
    const _Float16* arow = &Ash[mTile * 16 + l15][0];
    const _Float16* brow = &BshT[nTile * 16 + l15][0];
    v8f acc = {};
#pragma unroll
    for (int kc = 0; kc < C_; kc += 32) {
        v16h a  = frag_a(arow, kc, lane);
        v16h bf = frag_b(brow, kc, lane);
        acc = __builtin_amdgcn_wmma_f32_16x16x32_f16(
            false, a, false, bf, (short)0, acc, false, false);
    }

#if defined(HAVE_GFX1250_ASYNC)
    __builtin_amdgcn_s_wait_asynccnt(0);   // this wave's async copies done
#endif
    __syncthreads();                        // all waves' copies done

    // C/D layout: lanes 0-15 -> N=lane, M=r; lanes 16-31 -> N=lane-16, M=8+r
    int n = nTile * 16 + l15;
    float bval = bias[n];
#pragma unroll
    for (int r = 0; r < 8; r++) {
        int m = mTile * 16 + ((lane < 16) ? r : (8 + r));
        float v = acc[r] + bval + Flds[m][n];
        hout[(size_t)(pbase + m) * C_ + n] = gelu_approx(v);
    }
}

// ---------------------------------------------------------------- fused projection
__global__ __launch_bounds__(256) void k_proj(
        const float* __restrict__ hin,
        const float* __restrict__ w1, const float* __restrict__ b1,
        const float* __restrict__ w2, const float* __restrict__ b2,
        float* __restrict__ out) {
    __shared__ __align__(32) _Float16 Ash[32][C_];     // [M][K]
    __shared__ __align__(32) _Float16 BshT[PH_][C_];   // [N][K] transposed
    __shared__ float Hid[32][PH_];
    int tid = threadIdx.x;
    int pbase = blockIdx.x * 32;

    for (int i = tid; i < (32 * C_) / 4; i += 256) {
        int m = (i * 4) / C_, k0 = (i * 4) % C_;
        float4 v = *(const float4*)(hin + (size_t)(pbase + m) * C_ + k0);
        v4h h4 = { (_Float16)v.x, (_Float16)v.y, (_Float16)v.z, (_Float16)v.w };
        *(v4h*)&Ash[m][k0] = h4;
    }
    for (int i = tid; i < C_ * PH_; i += 256) {
        int k = i / PH_, n = i % PH_;
        BshT[n][k] = (_Float16)w1[i];
    }
    __syncthreads();

    int wave = tid >> 5, lane = tid & 31;
    int l15 = lane & 15;
#pragma unroll
    for (int t2 = 0; t2 < 2; t2++) {
        int tile = wave + 8 * t2;          // 2 M-tiles x 8 N-tiles
        int mTile = tile >> 3, nTile = tile & 7;
        const _Float16* arow = &Ash[mTile * 16 + l15][0];
        const _Float16* brow = &BshT[nTile * 16 + l15][0];
        v8f acc = {};
#pragma unroll
        for (int kc = 0; kc < C_; kc += 32) {
            v16h a  = frag_a(arow, kc, lane);
            v16h bf = frag_b(brow, kc, lane);
            acc = __builtin_amdgcn_wmma_f32_16x16x32_f16(
                false, a, false, bf, (short)0, acc, false, false);
        }
        int n = nTile * 16 + l15;
        float bb = b1[n];
#pragma unroll
        for (int r = 0; r < 8; r++) {
            int m = mTile * 16 + ((lane < 16) ? r : (8 + r));
            Hid[m][n] = gelu_approx(acc[r] + bb);
        }
    }
    __syncthreads();
    if (tid < 64) {
        int m = tid >> 1, j = tid & 1;
        float s = b2[j];
        for (int k = 0; k < PH_; k++) s += Hid[m][k] * w2[k * 2 + j];
        out[(size_t)(pbase + m) * 2 + j] = s;   // complex64 interleaved (re, im)
    }
}

// ---------------------------------------------------------------- host launcher
extern "C" void kernel_launch(void* const* d_in, const int* in_sizes, int n_in,
                              void* d_out, int out_size, void* d_ws, size_t ws_size,
                              hipStream_t stream) {
    (void)in_sizes; (void)n_in; (void)out_size; (void)ws_size;
    const float* x       = (const float*)d_in[0];
    const float* lift_w  = (const float*)d_in[1];
    const float* lift_b  = (const float*)d_in[2];
    const float* kernels = (const float*)d_in[3];  // complex64 interleaved
    const float* conv_ws = (const float*)d_in[4];
    const float* conv_bs = (const float*)d_in[5];
    const float* proj1_w = (const float*)d_in[6];
    const float* proj1_b = (const float*)d_in[7];
    const float* proj2_w = (const float*)d_in[8];
    const float* proj2_b = (const float*)d_in[9];
    float* out = (float*)d_out;

    float* ws = (float*)d_ws;
    size_t off = 0;
    float* twc = ws + off; off += (size_t)KMAX_ * H_;
    float* tws = ws + off; off += (size_t)KMAX_ * H_;
    float* hA  = ws + off; off += (size_t)NPIX_ * C_;
    float* hB  = ws + off; off += (size_t)NPIX_ * C_;
    float* Rre = ws + off; off += (size_t)B_ * KMAX_ * W_ * C_;
    float* Rim = ws + off; off += (size_t)B_ * KMAX_ * W_ * C_;
    float* Lre = ws + off; off += (size_t)B_ * KMAX_ * KMAX_ * C_;
    float* Lim = ws + off; off += (size_t)B_ * KMAX_ * KMAX_ * C_;
    float* Mre = ws + off; off += (size_t)B_ * KMAX_ * KMAX_ * C_;
    float* Mim = ws + off; off += (size_t)B_ * KMAX_ * KMAX_ * C_;
    float* Zre = Rre;   // R dead after k_coldft; Z aliases R (same size)
    float* Zim = Rim;

    k_twiddle<<<(KMAX_ * H_ + 255) / 256, 256, 0, stream>>>(twc, tws);
    k_lift<<<((size_t)NPIX_ * C_) / 256, 256, 0, stream>>>(x, lift_w, lift_b, hA);

    float* cur = hA;
    float* nxt = hB;
    for (int s = 0; s < STAGES_; s++) {
        const float* kern = kernels + (size_t)s * C_ * C_ * KMAX_ * KMAX_ * 2;
        const float* cw   = conv_ws + (size_t)s * C_ * C_;
        const float* cb   = conv_bs + (size_t)s * C_;

        k_rowdft<<<(B_ * W_ * C_) / 256, 256, 0, stream>>>(cur, twc, tws, Rre, Rim);
        k_coldft<<<(B_ * KMAX_ * KMAX_ * C_) / 256, 256, 0, stream>>>(Rre, Rim, twc, tws, Lre, Lim);
        k_mix<<<(B_ * KMAX_ * KMAX_ * C_) / 256, 256, 0, stream>>>(Lre, Lim, kern, Mre, Mim);
        k_irow<<<(B_ * H_ * KMAX_ * C_) / 256, 256, 0, stream>>>(Mre, Mim, twc, tws, Zre, Zim);
        k_icol<<<((size_t)NPIX_ * C_) / 256, 256, 0, stream>>>(Zre, Zim, twc, tws, nxt);
        k_conv_wmma<<<NPIX_ / 32, 256, 0, stream>>>(cur, nxt, cw, cb, nxt);

        float* tmp = cur; cur = nxt; nxt = tmp;
    }

    k_proj<<<NPIX_ / 32, 256, 0, stream>>>(cur, proj1_w, proj1_b, proj2_w, proj2_b, out);
}
